// ResidualGATBlock_89644557402834
// MI455X (gfx1250) — matrix operand order repaired and checked by reading, not compiled, not run on hardware
//
#include <hip/hip_runtime.h>
#include <hip/hip_bf16.h>
#include <math.h>

#define NEG_SLOPE 0.2f
#define LN_EPS 1e-5f

typedef __attribute__((ext_vector_type(16))) __bf16 v16bf;
typedef __attribute__((ext_vector_type(8)))  float  v8f;

union Frag16 {
  v16bf v;
  unsigned u[8];
  uint4 q[2];
};

__device__ __forceinline__ unsigned short f2bf(float f) {
  union { __bf16 h; unsigned short s; } c;
  c.h = (__bf16)f;
  return c.s;
}
__device__ __forceinline__ unsigned pk2(float lo, float hi) {
  union { __bf16 h[2]; unsigned u; } c;
  c.h[0] = (__bf16)lo;
  c.h[1] = (__bf16)hi;
  return c.u;
}
// order-preserving mapping float -> u32 for atomic max
__device__ __forceinline__ unsigned fenc(float f) {
  int i = __float_as_int(f);
  return (i >= 0) ? ((unsigned)i | 0x80000000u) : (unsigned)(~i);
}
__device__ __forceinline__ float fdec(unsigned u) {
  int i = (u & 0x80000000u) ? (int)(u & 0x7fffffffu) : (int)(~u);
  return __int_as_float(i);
}

// -------- init: zero output accumulator, seed max (-inf) and Z --------
__global__ void init_k(float* __restrict__ out, unsigned* __restrict__ mbuf,
                       float* __restrict__ zbuf, int N) {
  int idx = blockIdx.x * blockDim.x + threadIdx.x;
  if (idx < N * 128) out[idx] = 0.f;
  if (idx < N * 8) { mbuf[idx] = 0x007fffffu; /* enc(-inf) */ zbuf[idx] = 0.f; }
}

// -------- h = x @ W + b  via v_wmma_f32_16x16x32_bf16 --------
// block = 256 thr (8 waves); block tile = 128 rows x 128 cols; wave = 16 rows.
// LDS holds W^T bf16 with stride 136 halves (272B): 16B-aligned rows, bank-safe.
#define LWS 136
__global__ __launch_bounds__(256)
void gat_gemm_wmma(const float* __restrict__ x, const float* __restrict__ W,
                   const float* __restrict__ bias, float* __restrict__ hout, int N) {
  __shared__ unsigned short lw[128 * LWS];
  const int tid = threadIdx.x;
  const int nbase = blockIdx.x * 128;

  // stage W^T: lw[j*LWS + k] = bf16(W[k*128 + j]); global reads coalesced over j
  #pragma unroll 4
  for (int i = 0; i < 64; ++i) {
    int idx = tid + i * 256;
    int k = idx >> 7, j = idx & 127;
    lw[j * LWS + k] = f2bf(W[k * 128 + j]);
  }
  __syncthreads();

  const int lane = tid & 31;
  const int wid  = tid >> 5;
  const int hi   = (lane >= 16) ? 1 : 0;
  const int mrow = lane & 15;
  const int koff = hi ? 8 : 0;       // A-frag K offset for hi half-wave
  const int kbB  = hi ? 8 : 0;       // B-frag K offset for hi half-wave

  // preload this wave's 16-row A tile as 4 bf16 fragments (K-steps of 32)
  const int grow = nbase + wid * 16 + mrow;
  const bool ok = grow < N;
  Frag16 afr[4];
  const float4 zf = {0.f, 0.f, 0.f, 0.f};
  #pragma unroll
  for (int kt = 0; kt < 4; ++kt) {
    int b0 = kt * 32 + koff;
    float4 p0 = ok ? *(const float4*)(x + (size_t)grow * 128 + b0)      : zf;
    float4 p1 = ok ? *(const float4*)(x + (size_t)grow * 128 + b0 + 4)  : zf;
    float4 p2 = ok ? *(const float4*)(x + (size_t)grow * 128 + b0 + 16) : zf;
    float4 p3 = ok ? *(const float4*)(x + (size_t)grow * 128 + b0 + 20) : zf;
    afr[kt].u[0] = pk2(p0.x, p0.y); afr[kt].u[1] = pk2(p0.z, p0.w);
    afr[kt].u[2] = pk2(p1.x, p1.y); afr[kt].u[3] = pk2(p1.z, p1.w);
    afr[kt].u[4] = pk2(p2.x, p2.y); afr[kt].u[5] = pk2(p2.z, p2.w);
    afr[kt].u[6] = pk2(p3.x, p3.y); afr[kt].u[7] = pk2(p3.z, p3.w);
  }

  v8f acc[8];
  const v8f vz = {0.f, 0.f, 0.f, 0.f, 0.f, 0.f, 0.f, 0.f};
  #pragma unroll
  for (int ct = 0; ct < 8; ++ct) acc[ct] = vz;

  #pragma unroll
  for (int kt = 0; kt < 4; ++kt) {
    const int kb = kt * 32;
    // preload all 8 B fragments for this K-step (16x ds_load_b128 in flight)
    Frag16 bm[8];
    #pragma unroll
    for (int ct = 0; ct < 8; ++ct) {
      const uint4* pb =
          (const uint4*)(lw + (ct * 16 + mrow) * LWS + kb + kbB);
      bm[ct].q[0] = pb[0];
      bm[ct].q[1] = pb[2];
    }
    // 8 independent WMMAs back-to-back (shared A, disjoint B/D)
    #pragma unroll
    for (int ct = 0; ct < 8; ++ct) {
      acc[ct] = __builtin_amdgcn_wmma_f32_16x16x32_bf16(
          false, afr[kt].v, false, bm[ct].v, (short)0, acc[ct], false, false);
    }
  }

  // store D (+bias): VGPR r, lanes 0-15 -> M=r, lanes 16-31 -> M=r+8
  #pragma unroll
  for (int ct = 0; ct < 8; ++ct) {
    const int col = ct * 16 + mrow;
    const float bv = bias[col];
    #pragma unroll
    for (int r = 0; r < 8; ++r) {
      int orow = nbase + wid * 16 + (hi ? r + 8 : r);
      if (orow < N) hout[(size_t)orow * 128 + col] = acc[ct][r] + bv;
    }
  }
}

// -------- per-node attention dots: a_src[n,h], a_dst[n,h] --------
__global__ void node_att_k(const float* __restrict__ h, const float* __restrict__ as,
                           const float* __restrict__ ad, float* __restrict__ oas,
                           float* __restrict__ oad, int N) {
  int idx = blockIdx.x * blockDim.x + threadIdx.x;
  if (idx >= N * 8) return;
  int n = idx >> 3, hd = idx & 7;
  const float4* h4 = (const float4*)(h + (size_t)n * 128 + hd * 16);
  const float4* s4 = (const float4*)(as + hd * 16);
  const float4* d4 = (const float4*)(ad + hd * 16);
  float ss = 0.f, ds = 0.f;
  #pragma unroll
  for (int i = 0; i < 4; ++i) {
    float4 hv = h4[i], sv = s4[i], dv = d4[i];
    ss += hv.x * sv.x + hv.y * sv.y + hv.z * sv.z + hv.w * sv.w;
    ds += hv.x * dv.x + hv.y * dv.y + hv.z * dv.z + hv.w * dv.w;
  }
  oas[idx] = ss;
  oad[idx] = ds;
}

// -------- EDGE_DIM==1 collapse: a_edge[e,h] = ea[e]*we[h] + be[h] --------
__global__ void edge_coef_k(const float* __restrict__ lw, const float* __restrict__ lb,
                            const float* __restrict__ ae, float* __restrict__ webe) {
  int hd = threadIdx.x;
  if (hd >= 8) return;
  float w = 0.f, b = 0.f;
  for (int c = 0; c < 16; ++c) {
    float a = ae[hd * 16 + c];
    w += lw[hd * 16 + c] * a;
    b += lb[hd * 16 + c] * a;
  }
  webe[hd] = w;
  webe[8 + hd] = b;
}

// -------- segment max over dst: one thread per edge, 8 heads, wide loads --------
__global__ void edge_max_k(const int* __restrict__ src, const int* __restrict__ dst,
                           const float* __restrict__ ea, const float* __restrict__ asrc,
                           const float* __restrict__ adst, const float* __restrict__ webe,
                           unsigned* __restrict__ mbuf, int E) {
  int e = blockIdx.x * blockDim.x + threadIdx.x;
  if (e >= E) return;
  int s = src[e], d = dst[e];
  float eav = ea[e];
  float4 s0 = ((const float4*)asrc)[s * 2], s1 = ((const float4*)asrc)[s * 2 + 1];
  float4 d0 = ((const float4*)adst)[d * 2], d1 = ((const float4*)adst)[d * 2 + 1];
  float av[8] = {s0.x + d0.x, s0.y + d0.y, s0.z + d0.z, s0.w + d0.w,
                 s1.x + d1.x, s1.y + d1.y, s1.z + d1.z, s1.w + d1.w};
  #pragma unroll
  for (int hd = 0; hd < 8; ++hd) {
    float l = av[hd] + eav * webe[hd] + webe[8 + hd];
    l = l > 0.f ? l : NEG_SLOPE * l;
    atomicMax(&mbuf[d * 8 + hd], fenc(l));
  }
}

// -------- segment sum of exp(logit - max): one thread per edge --------
__global__ void edge_sum_k(const int* __restrict__ src, const int* __restrict__ dst,
                           const float* __restrict__ ea, const float* __restrict__ asrc,
                           const float* __restrict__ adst, const float* __restrict__ webe,
                           const unsigned* __restrict__ mbuf, float* __restrict__ zbuf,
                           int E) {
  int e = blockIdx.x * blockDim.x + threadIdx.x;
  if (e >= E) return;
  int s = src[e], d = dst[e];
  float eav = ea[e];
  float4 s0 = ((const float4*)asrc)[s * 2], s1 = ((const float4*)asrc)[s * 2 + 1];
  float4 d0 = ((const float4*)adst)[d * 2], d1 = ((const float4*)adst)[d * 2 + 1];
  uint4 m0 = ((const uint4*)mbuf)[d * 2], m1 = ((const uint4*)mbuf)[d * 2 + 1];
  float av[8] = {s0.x + d0.x, s0.y + d0.y, s0.z + d0.z, s0.w + d0.w,
                 s1.x + d1.x, s1.y + d1.y, s1.z + d1.z, s1.w + d1.w};
  unsigned mv[8] = {m0.x, m0.y, m0.z, m0.w, m1.x, m1.y, m1.z, m1.w};
  #pragma unroll
  for (int hd = 0; hd < 8; ++hd) {
    float l = av[hd] + eav * webe[hd] + webe[8 + hd];
    l = l > 0.f ? l : NEG_SLOPE * l;
    atomicAdd(&zbuf[d * 8 + hd], __expf(l - fdec(mv[hd])));
  }
}

// -------- scatter: out[dst] += alpha * h[src]; one wave per edge, 4 ch/lane --------
__global__ void edge_scatter_k(const int* __restrict__ src, const int* __restrict__ dst,
                               const float* __restrict__ ea, const float* __restrict__ asrc,
                               const float* __restrict__ adst, const float* __restrict__ webe,
                               const unsigned* __restrict__ mbuf, const float* __restrict__ zbuf,
                               const float* __restrict__ h, float* __restrict__ out, int E) {
  int idx = blockIdx.x * blockDim.x + threadIdx.x;
  if (idx >= E * 32) return;
  int e = idx >> 5, q = idx & 31;     // q = channel quad index; head = q/4
  int hd = q >> 2;
  int s = src[e], d = dst[e];
  float l = asrc[s * 8 + hd] + adst[d * 8 + hd] + ea[e] * webe[hd] + webe[8 + hd];
  l = l > 0.f ? l : NEG_SLOPE * l;
  float m = fdec(mbuf[d * 8 + hd]);
  float z = zbuf[d * 8 + hd];
  float alpha = __expf(l - m) / (z + 1e-16f);
  float4 hv = ((const float4*)h)[(size_t)s * 32 + q];
  float* ob = out + (size_t)d * 128 + q * 4;
  atomicAdd(ob + 0, alpha * hv.x);
  atomicAdd(ob + 1, alpha * hv.y);
  atomicAdd(ob + 2, alpha * hv.z);
  atomicAdd(ob + 3, alpha * hv.w);
}

// -------- fused +b +x residual, LayerNorm(128), ELU; one wave per node --------
__global__ __launch_bounds__(256)
void gat_finalize_k(const float* __restrict__ x, const float* __restrict__ bias,
                    const float* __restrict__ gamma, const float* __restrict__ beta,
                    float* __restrict__ out, int N) {
  int wid = threadIdx.x >> 5;
  int lane = threadIdx.x & 31;
  int n = blockIdx.x * 8 + wid;
  if (n >= N) return;
  float4 ov = ((const float4*)(out + (size_t)n * 128))[lane];
  float4 xv = ((const float4*)(x + (size_t)n * 128))[lane];
  float4 bv = ((const float4*)bias)[lane];
  float v0 = ov.x + bv.x + xv.x;
  float v1 = ov.y + bv.y + xv.y;
  float v2 = ov.z + bv.z + xv.z;
  float v3 = ov.w + bv.w + xv.w;
  float s = v0 + v1 + v2 + v3;
  #pragma unroll
  for (int off = 16; off > 0; off >>= 1) s += __shfl_xor(s, off, 32);
  float mu = s * (1.f / 128.f);
  float d0 = v0 - mu, d1 = v1 - mu, d2 = v2 - mu, d3 = v3 - mu;
  float q = d0 * d0 + d1 * d1 + d2 * d2 + d3 * d3;
  #pragma unroll
  for (int off = 16; off > 0; off >>= 1) q += __shfl_xor(q, off, 32);
  float inv = rsqrtf(q * (1.f / 128.f) + LN_EPS);
  float4 gv = ((const float4*)gamma)[lane];
  float4 tv = ((const float4*)beta)[lane];
  float o0 = d0 * inv * gv.x + tv.x;
  float o1 = d1 * inv * gv.y + tv.y;
  float o2 = d2 * inv * gv.z + tv.z;
  float o3 = d3 * inv * gv.w + tv.w;
  o0 = o0 > 0.f ? o0 : expm1f(o0);
  o1 = o1 > 0.f ? o1 : expm1f(o1);
  o2 = o2 > 0.f ? o2 : expm1f(o2);
  o3 = o3 > 0.f ? o3 : expm1f(o3);
  float4 res = {o0, o1, o2, o3};
  ((float4*)(out + (size_t)n * 128))[lane] = res;
}

extern "C" void kernel_launch(void* const* d_in, const int* in_sizes, int n_in,
                              void* d_out, int out_size, void* d_ws, size_t ws_size,
                              hipStream_t stream) {
  (void)n_in; (void)out_size; (void)ws_size;
  const float* x        = (const float*)d_in[0];
  const int*   ei       = (const int*)d_in[1];
  const float* ea       = (const float*)d_in[2];
  const float* W        = (const float*)d_in[3];
  const float* b        = (const float*)d_in[4];
  const float* att_src  = (const float*)d_in[5];
  const float* att_dst  = (const float*)d_in[6];
  const float* att_edge = (const float*)d_in[7];
  const float* lew      = (const float*)d_in[8];
  const float* leb      = (const float*)d_in[9];
  const float* gamma    = (const float*)d_in[10];
  const float* beta     = (const float*)d_in[11];
  float* out = (float*)d_out;

  const int N = in_sizes[0] / 128;
  const int E = in_sizes[1] / 2;
  const int* src = ei;
  const int* dst = ei + E;

  // workspace carve-up (~32 MB for N=50000)
  float* ws = (float*)d_ws;
  size_t off = 0;
  float* h    = ws + off; off += (size_t)N * 128;
  float* asrc = ws + off; off += (size_t)N * 8;
  float* adst = ws + off; off += (size_t)N * 8;
  float* webe = ws + off; off += 16;
  unsigned* mbuf = (unsigned*)(ws + off); off += (size_t)N * 8;
  float* zbuf = ws + off; off += (size_t)N * 8;

  init_k<<<(N * 128 + 255) / 256, 256, 0, stream>>>(out, mbuf, zbuf, N);
  gat_gemm_wmma<<<(N + 127) / 128, 256, 0, stream>>>(x, W, b, h, N);
  node_att_k<<<(N * 8 + 255) / 256, 256, 0, stream>>>(h, att_src, att_dst, asrc, adst, N);
  edge_coef_k<<<1, 32, 0, stream>>>(lew, leb, att_edge, webe);
  edge_max_k<<<(E + 255) / 256, 256, 0, stream>>>(src, dst, ea, asrc, adst, webe, mbuf, E);
  edge_sum_k<<<(E + 255) / 256, 256, 0, stream>>>(src, dst, ea, asrc, adst, webe, mbuf, zbuf, E);
  edge_scatter_k<<<(E * 32 + 255) / 256, 256, 0, stream>>>(src, dst, ea, asrc, adst, webe,
                                                           mbuf, zbuf, h, out, E);
  gat_finalize_k<<<(N + 7) / 8, 256, 0, stream>>>(x, b, gamma, beta, out, N);
}
